// BRIMCell_78915729097548
// MI455X (gfx1250) — matrix operand order repaired and checked by reading, not compiled, not run on hardware
//
#include <hip/hip_runtime.h>
#include <math.h>

#define BATCH 8192
#define NBLK  8
#define HID   256
#define DIM   256
#define KACT  4
#define DKIN  16
#define DVIN  32
#define NHC   4
#define DKC   8
#define DVC   256

typedef _Float16 v16h __attribute__((ext_vector_type(16)));
typedef _Float16 v8h  __attribute__((ext_vector_type(8)));
typedef float    v8f  __attribute__((ext_vector_type(8)));

__device__ __forceinline__ v8f wmma_f16(v16h a, v16h b, v8f c) {
  // D = A(16x32 f16) * B(32x16 f16) + C(16x16 f32)
  return __builtin_amdgcn_wmma_f32_16x16x32_f16(false, a, false, b, (short)0, c, false, false);
}

// A fragment, row-major A, K contiguous. Lane (g=lane>>4, m=lane&15) holds
// row m, K [16g,16g+16) contiguous -> one 32B load.
__device__ __forceinline__ v16h load_a(const _Float16* __restrict__ row, int g) {
  return *(const v16h*)(row + 16 * g);
}
// B fragment from pre-transposed weights WT[col][k]: lane holds column c,
// K [16g,16g+16) contiguous -> one 32B load.
__device__ __forceinline__ v16h load_bT(const _Float16* __restrict__ colrow, int g) {
  return *(const v16h*)(colrow + 16 * g);
}

// ---------------- K0a: f32 -> f16 convert (grid-stride) ----------------
__global__ __launch_bounds__(256)
void k0_cvt(const float* __restrict__ src, _Float16* __restrict__ dst, size_t n) {
  for (size_t i = (size_t)blockIdx.x * blockDim.x + threadIdx.x; i < n;
       i += (size_t)gridDim.x * blockDim.x)
    dst[i] = (_Float16)src[i];
}
// ---------------- K0b: f32 (N,K,O) -> f16 transposed (N,O,K) ----------------
__global__ __launch_bounds__(256)
void k0_cvtT(const float* __restrict__ src, _Float16* __restrict__ dst,
             int K, int O, size_t total) {
  for (size_t i = (size_t)blockIdx.x * blockDim.x + threadIdx.x; i < total;
       i += (size_t)gridDim.x * blockDim.x) {
    const size_t n = i / ((size_t)K * O);
    const size_t r = i - n * (size_t)K * O;
    const int o = (int)(r / K), k = (int)(r - (size_t)o * K);
    dst[i] = (_Float16)src[n * (size_t)K * O + (size_t)k * O + o];
  }
}

// ---------------- K1: input attention, top-k mask, "inputs" ----------------
__global__ __launch_bounds__(256)
void k1_prelim(const float* __restrict__ x1, const float* __restrict__ x2,
               const float* __restrict__ hs,
               const float* __restrict__ Wkey, const float* __restrict__ bkey,
               const float* __restrict__ Wval, const float* __restrict__ bval,
               const float* __restrict__ Wq,
               _Float16* __restrict__ in_h, float* __restrict__ maskp) {
  __shared__ float sKL[8][48];
  __shared__ float sVL[8][96];
  __shared__ float sQL[8][128];
  __shared__ float sSC[8][24];
  __shared__ float sPR[8][24];
  __shared__ float sMK[8][8];
  const int wid = threadIdx.x >> 5, lane = threadIdx.x & 31;
  const int b = blockIdx.x * 8 + wid;
  {  // kL slots 0,1 (slot 2 = null input -> bkey)
    const int s = lane >> 4, kk = lane & 15;
    const float* xs = (s ? x2 : x1) + (size_t)b * DIM;
    float acc = 0.f;
    for (int d = 0; d < DIM; ++d) acc += xs[d] * Wkey[d * DKIN + kk];
    sKL[wid][s * 16 + kk] = acc + bkey[kk];
    if (lane < 16) sKL[wid][32 + lane] = bkey[lane];
  }
  for (int t = 0; t < 2; ++t) {  // vL slots 0,1
    const int i = lane + 32 * t, s = i >> 5, j = i & 31;
    const float* xs = (s ? x2 : x1) + (size_t)b * DIM;
    float acc = 0.f;
    for (int d = 0; d < DIM; ++d) acc += xs[d] * Wval[d * DVIN + j];
    sVL[wid][s * 32 + j] = acc + bval[j];
  }
  sVL[wid][64 + lane] = bval[lane];  // null slot
  for (int t = 0; t < 4; ++t) {      // qL: 8 blocks x 16
    const int i = lane + 32 * t, n = i >> 4, kk = i & 15;
    const float* hv = hs + (size_t)b * (NBLK * HID) + n * HID;
    const float* W = Wq + (size_t)n * (HID * DKIN) + kk;
    float acc = 0.f;
    for (int h = 0; h < HID; ++h) acc += hv[h] * W[h * DKIN];
    sQL[wid][n * 16 + kk] = acc;
  }
  __syncthreads();
  if (lane < 24) {  // scores[n][s] = q.k / sqrt(16)
    const int n = lane / 3, s = lane % 3;
    float acc = 0.f;
    for (int kk = 0; kk < 16; ++kk) acc += sQL[wid][n * 16 + kk] * sKL[wid][s * 16 + kk];
    sSC[wid][n * 3 + s] = acc * 0.25f;
  }
  __syncthreads();
  if (lane == 0) {  // top-(N-KACT) null scores -> mask 0 (jax tie-break: lowest idx)
    float ns[8]; bool sel[8];
    for (int n = 0; n < 8; ++n) { ns[n] = sSC[wid][n * 3 + 2]; sel[n] = false; }
    for (int t = 0; t < NBLK - KACT; ++t) {
      int best = 0; float bv = -3.4e38f;
      for (int n = 0; n < 8; ++n)
        if (!sel[n] && ns[n] > bv) { bv = ns[n]; best = n; }
      sel[best] = true;
    }
    for (int n = 0; n < 8; ++n) sMK[wid][n] = sel[n] ? 0.f : 1.f;
  }
  if (lane < 8) {  // softmax over 3 slots
    const int n = lane;
    const float a = sSC[wid][n * 3], bb = sSC[wid][n * 3 + 1], cc = sSC[wid][n * 3 + 2];
    const float mx = fmaxf(a, fmaxf(bb, cc));
    const float ea = expf(a - mx), eb = expf(bb - mx), ec = expf(cc - mx);
    const float den = ea + eb + ec;
    sPR[wid][n * 3] = ea / den; sPR[wid][n * 3 + 1] = eb / den; sPR[wid][n * 3 + 2] = ec / den;
  }
  __syncthreads();
  for (int t = 0; t < 8; ++t) {  // inputs = (probs @ vL) * mask -> f16
    const int i = lane + 32 * t, n = i >> 5, dd = i & 31;
    float v = 0.f;
    for (int s = 0; s < 3; ++s) v += sPR[wid][n * 3 + s] * sVL[wid][s * 32 + dd];
    v *= sMK[wid][n];
    in_h[(size_t)b * (NBLK * DVIN) + n * DVIN + dd] = (_Float16)v;
  }
  if (lane < 8) maskp[(size_t)b * NBLK + lane] = sMK[wid][lane];
}

// ---------------- K2: gate_h (WMMA K=256) + gate_x (WMMA K=32) + GRU ----------------
// 8 waves/block, each wave owns a 16-batch x 16-h tile for block n.
// A fragment shared across the 3 gate B-tiles. No LDS: direct b128 operand loads.
__global__ __launch_bounds__(256)
void k2_gates_gru(const _Float16* __restrict__ hs_h, const _Float16* __restrict__ in_h,
                  const _Float16* __restrict__ wh2hT, const _Float16* __restrict__ wx2hT,
                  const float* __restrict__ hs, float* __restrict__ hsnew,
                  _Float16* __restrict__ hsnew_h) {
  const int wid = threadIdx.x >> 5, lane = threadIdx.x & 31;
  const int b0 = (blockIdx.x * 8 + wid) * 16;
  const int o0 = blockIdx.y * 16, n = blockIdx.z;
  const int g = lane >> 4, c = lane & 15;
  v8f accR = {}, accZ = {}, accHN = {}, accIN = {};
  const _Float16* Arow = hs_h + (size_t)(b0 + c) * (NBLK * HID) + n * HID;
  const _Float16* Wb = wh2hT + (size_t)n * (3 * HID * HID);
  const _Float16* bR = Wb + (size_t)(0 * HID + o0 + c) * HID;
  const _Float16* bZ = Wb + (size_t)(1 * HID + o0 + c) * HID;
  const _Float16* bN = Wb + (size_t)(2 * HID + o0 + c) * HID;
#pragma unroll
  for (int k0 = 0; k0 < HID; k0 += 32) {
    if (k0 + 32 < HID) __builtin_prefetch(Arow + k0 + 32, 0, 1);
    const v16h a = load_a(Arow + k0, g);
    accR  = wmma_f16(a, load_bT(bR + k0, g), accR);
    accZ  = wmma_f16(a, load_bT(bZ + k0, g), accZ);
    accHN = wmma_f16(a, load_bT(bN + k0, g), accHN);
  }
  {  // gate_x: K=32, one WMMA per gate
    const _Float16* Ax = in_h + (size_t)(b0 + c) * (NBLK * DVIN) + n * DVIN;
    const _Float16* Wx = wx2hT + (size_t)n * (DVIN * 3 * HID);
    const v16h a = load_a(Ax, g);
    accR  = wmma_f16(a, load_bT(Wx + (size_t)(0 * HID + o0 + c) * DVIN, g), accR);
    accZ  = wmma_f16(a, load_bT(Wx + (size_t)(1 * HID + o0 + c) * DVIN, g), accZ);
    accIN = wmma_f16(a, load_bT(Wx + (size_t)(2 * HID + o0 + c) * DVIN, g), accIN);
  }
#pragma unroll
  for (int e = 0; e < 8; ++e) {  // GRU epilogue; C/D layout: row e+8g, col c
    const int bb = b0 + e + 8 * g;
    const size_t idx = (size_t)bb * (NBLK * HID) + n * HID + o0 + c;
    const float r = 1.f / (1.f + expf(-accR[e]));
    const float z = 1.f / (1.f + expf(-accZ[e]));
    const float nn = tanhf(accIN[e] + r * accHN[e]);
    const float hold = hs[idx];
    const float hnew = nn + z * (hold - nn);
    hsnew[idx] = hnew;
    hsnew_h[idx] = (_Float16)hnew;
  }
}

// ---------------- K3: q2/k2 (VALU K=256), 8x8 block attention + mask ----------------
__global__ __launch_bounds__(256)
void k3_att(const float* __restrict__ hsnew, const float* __restrict__ Wq_,
            const float* __restrict__ Wk_, const float* __restrict__ maskp,
            float* __restrict__ attp) {
  __shared__ float sQK[8][512];
  __shared__ float sS[8][256];
  const int wid = threadIdx.x >> 5, lane = threadIdx.x & 31;
  const int b = blockIdx.x * 8 + wid;
  for (int t = 0; t < 16; ++t) {
    const int i = lane + 32 * t;
    const int which = i >> 8, n = (i >> 5) & 7, kk = i & 31;
    const float* W = (which ? Wk_ : Wq_) + (size_t)n * (HID * 32) + kk;
    const float* hv = hsnew + (size_t)b * (NBLK * HID) + n * HID;
    float acc = 0.f;
    for (int h = 0; h < HID; ++h) acc += hv[h] * W[h * 32];
    sQK[wid][i] = acc;
  }
  __syncthreads();
  for (int t = 0; t < 8; ++t) {  // scores[h][n][m] = q2.k2 / sqrt(8)
    const int i = lane + 32 * t;
    const int h = i >> 6, n = (i >> 3) & 7, m = i & 7;
    float s = 0.f;
    for (int d = 0; d < DKC; ++d)
      s += sQK[wid][n * 32 + h * 8 + d] * sQK[wid][256 + m * 32 + h * 8 + d];
    sS[wid][i] = s * 0.3535533905932738f;
  }
  __syncthreads();
  {  // softmax over m, then row-mask
    const int h = lane >> 3, n = lane & 7;
    float mx = -3.4e38f;
    for (int m = 0; m < 8; ++m) mx = fmaxf(mx, sS[wid][h * 64 + n * 8 + m]);
    float ex[8], den = 0.f;
    for (int m = 0; m < 8; ++m) { ex[m] = expf(sS[wid][h * 64 + n * 8 + m] - mx); den += ex[m]; }
    const float mval = maskp[(size_t)b * NBLK + n];
    for (int m = 0; m < 8; ++m)
      attp[(((size_t)b * NHC + h) * NBLK + n) * NBLK + m] = ex[m] / den * mval;
  }
}

// ---------------- K4: v2 = hs_new @ Wv_ (WMMA, K=256, 4 col-tiles/wave) ----------------
__global__ __launch_bounds__(256)
void k4_v2(const _Float16* __restrict__ hsnew_h, const _Float16* __restrict__ wvT,
           _Float16* __restrict__ v2_h) {
  const int wid = threadIdx.x >> 5, lane = threadIdx.x & 31;
  const int b0 = (blockIdx.x * 8 + wid) * 16;
  const int col0 = blockIdx.y * 64, m = blockIdx.z;
  const int g = lane >> 4, c = lane & 15;
  const _Float16* Arow = hsnew_h + (size_t)(b0 + c) * (NBLK * HID) + m * HID;
  const _Float16* Bb = wvT + (size_t)m * (NHC * DVC * HID);
  const _Float16* Brow[4];
#pragma unroll
  for (int t = 0; t < 4; ++t) Brow[t] = Bb + (size_t)(col0 + t * 16 + c) * HID;
  v8f acc[4] = {};
#pragma unroll
  for (int k0 = 0; k0 < HID; k0 += 32) {
    if (k0 + 32 < HID) __builtin_prefetch(Arow + k0 + 32, 0, 1);
    const v16h a = load_a(Arow + k0, g);
#pragma unroll
    for (int t = 0; t < 4; ++t) acc[t] = wmma_f16(a, load_bT(Brow[t] + k0, g), acc[t]);
  }
#pragma unroll
  for (int t = 0; t < 4; ++t) {
    const int col = col0 + t * 16 + c, h = col >> 8, dv = col & 255;
#pragma unroll
    for (int e = 0; e < 8; ++e) {
      const int bb = b0 + e + 8 * g;
      v2_h[(((size_t)bb * NHC + h) * NBLK + m) * DVC + dv] = (_Float16)acc[t][e];
    }
  }
}

// ---------------- K5: ctx = att @ v2 (K=8, VALU) ----------------
__global__ __launch_bounds__(256)
void k5_ctx(const float* __restrict__ attp, const _Float16* __restrict__ v2_h,
            _Float16* __restrict__ ctx_h) {
  const size_t idx = (size_t)blockIdx.x * blockDim.x + threadIdx.x;
  const size_t total = (size_t)BATCH * NBLK * (NHC * DVC);
  if (idx >= total) return;
  const size_t b = idx >> 13;
  const int rem = (int)(idx & 8191);
  const int n = rem >> 10, d = rem & 1023, h = d >> 8, dv = d & 255;
  const float* ar = attp + (((size_t)b * NHC + h) * NBLK + n) * NBLK;
  const _Float16* vr = v2_h + ((size_t)b * NHC + h) * (NBLK * DVC) + dv;
  float acc = 0.f;
#pragma unroll
  for (int m = 0; m < 8; ++m) acc += ar[m] * (float)vr[m * DVC];
  ctx_h[idx] = (_Float16)acc;
}

// ---------------- K6: out = ctx @ Wout + h_b, masked blend (WMMA K=1024) ----------------
__global__ __launch_bounds__(256)
void k6_out(const _Float16* __restrict__ ctx_h, const _Float16* __restrict__ woutT,
            const float* __restrict__ hsnew, const float* __restrict__ maskp,
            const float* __restrict__ hs, float* __restrict__ out) {
  const int wid = threadIdx.x >> 5, lane = threadIdx.x & 31;
  const int b0 = (blockIdx.x * 8 + wid) * 16;
  const int o0 = blockIdx.y * 64, n = blockIdx.z;
  const int g = lane >> 4, c = lane & 15;
  const _Float16* Arow = ctx_h + (size_t)(b0 + c) * (NBLK * NHC * DVC) + n * (NHC * DVC);
  const _Float16* Bb = woutT + (size_t)n * (DVC * NHC * DVC);
  const _Float16* Brow[4];
#pragma unroll
  for (int t = 0; t < 4; ++t) Brow[t] = Bb + (size_t)(o0 + t * 16 + c) * (NHC * DVC);
  v8f acc[4] = {};
  for (int d0 = 0; d0 < NHC * DVC; d0 += 32) {
    if (d0 + 32 < NHC * DVC) __builtin_prefetch(Arow + d0 + 32, 0, 1);
    const v16h a = load_a(Arow + d0, g);
#pragma unroll
    for (int t = 0; t < 4; ++t) acc[t] = wmma_f16(a, load_bT(Brow[t] + d0, g), acc[t]);
  }
#pragma unroll
  for (int t = 0; t < 4; ++t) {
#pragma unroll
    for (int e = 0; e < 8; ++e) {
      const int bb = b0 + e + 8 * g;
      const size_t idx = (size_t)bb * (NBLK * HID) + n * HID + o0 + t * 16 + c;
      const float m = maskp[(size_t)bb * NBLK + n];
      const float val = acc[t][e] + hsnew[idx];
      out[idx] = m * val + (1.f - m) * hs[idx];
    }
  }
}

extern "C" void kernel_launch(void* const* d_in, const int* in_sizes, int n_in,
                              void* d_out, int out_size, void* d_ws, size_t ws_size,
                              hipStream_t stream) {
  (void)in_sizes; (void)n_in; (void)out_size; (void)ws_size;
  const float* x1   = (const float*)d_in[0];
  const float* x2   = (const float*)d_in[1];
  const float* hs   = (const float*)d_in[2];
  const float* Wkey = (const float*)d_in[3];
  const float* bkey = (const float*)d_in[4];
  const float* Wval = (const float*)d_in[5];
  const float* bval = (const float*)d_in[6];
  const float* Wq   = (const float*)d_in[7];
  const float* Wq_p = (const float*)d_in[8];
  const float* Wk_p = (const float*)d_in[9];
  const float* Wv_p = (const float*)d_in[10];
  const float* Wout = (const float*)d_in[11];
  const float* Wx2h = (const float*)d_in[12];
  const float* Wh2h = (const float*)d_in[13];
  float* out = (float*)d_out;

  char* ws = (char*)d_ws;
  size_t off = 0;
  auto alloc = [&](size_t bytes) -> void* {
    void* p = ws + off;
    off += (bytes + 255) & ~(size_t)255;
    return p;
  };
  const size_t nHS = (size_t)BATCH * NBLK * HID;
  _Float16* hs_h    = (_Float16*)alloc(nHS * 2);
  _Float16* wh2hT   = (_Float16*)alloc((size_t)NBLK * HID * 3 * HID * 2);   // (n, o, k)
  _Float16* wx2hT   = (_Float16*)alloc((size_t)NBLK * DVIN * 3 * HID * 2);  // (n, o, k)
  _Float16* wvT     = (_Float16*)alloc((size_t)NBLK * HID * NHC * DVC * 2); // (m, col, k)
  _Float16* woutT   = (_Float16*)alloc((size_t)NBLK * NHC * DVC * DVC * 2); // (n, o, d)
  _Float16* in_h    = (_Float16*)alloc((size_t)BATCH * NBLK * DVIN * 2);
  float*    maskp   = (float*)alloc((size_t)BATCH * NBLK * 4);
  float*    hsnew   = (float*)alloc(nHS * 4);
  _Float16* hsnew_h = (_Float16*)alloc(nHS * 2);
  float*    attp    = (float*)alloc((size_t)BATCH * NHC * NBLK * NBLK * 4);
  _Float16* v2_h    = (_Float16*)alloc((size_t)BATCH * NHC * NBLK * DVC * 2);
  _Float16* ctx_h   = (_Float16*)alloc((size_t)BATCH * NBLK * NHC * DVC * 2);

  k0_cvt<<<4096, 256, 0, stream>>>(hs, hs_h, nHS);
  k0_cvtT<<<1024, 256, 0, stream>>>(Wh2h, wh2hT, HID, 3 * HID,
                                    (size_t)NBLK * HID * 3 * HID);
  k0_cvtT<<<256, 256, 0, stream>>>(Wx2h, wx2hT, DVIN, 3 * HID,
                                   (size_t)NBLK * DVIN * 3 * HID);
  k0_cvtT<<<1024, 256, 0, stream>>>(Wv_p, wvT, HID, NHC * DVC,
                                    (size_t)NBLK * HID * NHC * DVC);
  k0_cvtT<<<1024, 256, 0, stream>>>(Wout, woutT, NHC * DVC, DVC,
                                    (size_t)NBLK * NHC * DVC * DVC);

  k1_prelim<<<BATCH / 8, 256, 0, stream>>>(x1, x2, hs, Wkey, bkey, Wval, bval, Wq, in_h, maskp);
  k2_gates_gru<<<dim3(BATCH / 128, HID / 16, NBLK), 256, 0, stream>>>(
      hs_h, in_h, wh2hT, wx2hT, hs, hsnew, hsnew_h);
  k3_att<<<BATCH / 8, 256, 0, stream>>>(hsnew, Wq_p, Wk_p, maskp, attp);
  k4_v2<<<dim3(BATCH / 128, (NHC * DVC) / 64, NBLK), 256, 0, stream>>>(hsnew_h, wvT, v2_h);
  {
    const size_t total = (size_t)BATCH * NBLK * NHC * DVC;
    k5_ctx<<<(unsigned)((total + 255) / 256), 256, 0, stream>>>(attp, v2_h, ctx_h);
  }
  k6_out<<<dim3(BATCH / 128, DVC / 64, NBLK), 256, 0, stream>>>(ctx_h, woutT, hsnew, maskp, hs, out);
}
